// TransitionModel_20418274526005
// MI455X (gfx1250) — matrix operand order repaired and checked by reading, not compiled, not run on hardware
//
#include <hip/hip_runtime.h>
#include <hip/hip_bf16.h>
#include <math.h>

#define NN 1024
#define BATCH 256

typedef __attribute__((ext_vector_type(16))) _Float16 v16h;
typedef __attribute__((ext_vector_type(8)))  _Float16 v8h;
typedef __attribute__((ext_vector_type(8)))  float    v8f;

// ---------------------------------------------------------------------------
// Kernel 1: per-column logsumexp of W  (log_softmax denominator over axis 0)
// lse[j] = max_i W[i,j] + log( sum_i exp(W[i,j] - max) )
// Threads own columns -> loads coalesced across the row dimension.
// ---------------------------------------------------------------------------
__global__ void col_lse_kernel(const float* __restrict__ W,
                               float* __restrict__ lse) {
    int col = blockIdx.x * blockDim.x + threadIdx.x;
    if (col >= NN) return;
    float m = -INFINITY;
    for (int row = 0; row < NN; ++row)
        m = fmaxf(m, W[row * NN + col]);
    float s = 0.0f;
    for (int row = 0; row < NN; ++row)
        s += expf(W[row * NN + col] - m);
    lse[col] = m + logf(s);
}

// ---------------------------------------------------------------------------
// Kernel 2: per-row max + exp-pack to f16.
//   v[j]        = X[row,j] - (use_lse ? lse[j] : 0)
//   rowmax[row] = max_j v[j]
//   packed[row,j] = (f16) exp(v[j] - rowmax)   in (0,1]
// One block (256 threads) per row; row cached in LDS between the two passes.
// ---------------------------------------------------------------------------
__global__ void rowpack_kernel(const float* __restrict__ X,
                               const float* __restrict__ lse, int use_lse,
                               float* __restrict__ rowmax,
                               _Float16* __restrict__ packed) {
    __shared__ float buf[NN];
    __shared__ float red[256];
    const int row = blockIdx.x;
    const int tid = threadIdx.x;

    float m = -INFINITY;
    #pragma unroll
    for (int t = 0; t < NN / 256; ++t) {
        int j = t * 256 + tid;
        float v = X[row * NN + j];
        if (use_lse) v -= lse[j];
        buf[j] = v;
        m = fmaxf(m, v);
    }
    red[tid] = m;
    __syncthreads();
    for (int off = 128; off > 0; off >>= 1) {
        if (tid < off) red[tid] = fmaxf(red[tid], red[tid + off]);
        __syncthreads();
    }
    m = red[0];
    if (tid == 0) rowmax[row] = m;

    #pragma unroll
    for (int t = 0; t < NN / 256; ++t) {
        int j = t * 256 + tid;
        packed[row * NN + j] = (_Float16)expf(buf[j] - m);
    }
}

// ---------------------------------------------------------------------------
// Kernel 3: WMMA GEMM  S[m,i] = sum_j Bexp[m,j] * Aexp[i,j]
// then fused epilogue   out[m,i] = Mb[m] + Ma[i] + log(S[m,i])
//
// D = A x B with  A-operand = Bexp tile (16 x 32, lane = batch row m),
//                 B-operand = Aexp tile (32 x 16, lane = output column i).
// Both operands are K(=j)-contiguous in memory -> b128 loads, no transpose.
// Each wave: 16(M) x 64(N) output tile, 4 f32 accumulators sharing one A frag.
// ---------------------------------------------------------------------------
__global__ __launch_bounds__(128)
void logmm_wmma_kernel(const _Float16* __restrict__ Aexp,  // [NN , NN]  row = i
                       const _Float16* __restrict__ Bexp,  // [BATCH, NN] row = m
                       const float*    __restrict__ Ma,    // [NN]
                       const float*    __restrict__ Mb,    // [BATCH]
                       float*          __restrict__ out) { // [BATCH, NN]
    const int lane = threadIdx.x & 31;
    const int wave = blockIdx.x * 4 + (threadIdx.x >> 5);
    const int NT   = NN / 64;                 // 16 column-tiles of 64
    const int mBase = (wave / NT) * 16;
    const int nBase = (wave % NT) * 64;
    const int half  = lane >> 4;              // 0: lanes 0-15, 1: lanes 16-31
    const int lr    = lane & 15;

    v8f acc[4] = {};

    for (int k0 = 0; k0 < NN; k0 += 32) {
        // --- A fragment: Bexp[mBase+lr, k0..k0+31], contiguous K order matching B
        // half=0: K 0-15 ; half=1: K 16-31
        v16h a;
        {
            const v8h* p0 = (const v8h*)(Bexp + (size_t)(mBase + lr) * NN + k0 + half * 16);
            v8h lo = p0[0], hi = p0[1];
            #pragma unroll
            for (int t = 0; t < 8; ++t) { a[t] = lo[t]; a[t + 8] = hi[t]; }
        }
        // --- 4 B fragments: Aexp[nBase+f*16+lr, k0 + half*16 .. +16]  (32 B contiguous)
        #pragma unroll
        for (int f = 0; f < 4; ++f) {
            v16h b;
            const v8h* q = (const v8h*)(Aexp + (size_t)(nBase + f * 16 + lr) * NN + k0 + half * 16);
            v8h lo = q[0], hi = q[1];
            #pragma unroll
            for (int t = 0; t < 8; ++t) { b[t] = lo[t]; b[t + 8] = hi[t]; }

            acc[f] = __builtin_amdgcn_wmma_f32_16x16x32_f16(
                /*neg_a=*/false, a, /*neg_b=*/false, b,
                /*c_mod=*/(short)0, acc[f], /*reuse_a=*/false, /*reuse_b=*/false);
        }
    }

    // Epilogue: C/D layout -> VGPR r, lane L: M = r + 8*(L>=16), N = L%16
    #pragma unroll
    for (int f = 0; f < 4; ++f) {
        const int i   = nBase + f * 16 + lr;
        const float mai = Ma[i];
        #pragma unroll
        for (int r = 0; r < 8; ++r) {
            const int m = mBase + r + half * 8;
            out[(size_t)m * NN + i] = Mb[m] + mai + logf(acc[f][r]);
        }
    }
}

// ---------------------------------------------------------------------------
extern "C" void kernel_launch(void* const* d_in, const int* in_sizes, int n_in,
                              void* d_out, int out_size, void* d_ws, size_t ws_size,
                              hipStream_t stream) {
    const float* log_alpha = (const float*)d_in[0];   // [BATCH, NN]
    const float* W         = (const float*)d_in[1];   // [NN, NN]
    float* out             = (float*)d_out;           // [BATCH, NN]

    // Workspace layout (all 16B-aligned):
    char* ws = (char*)d_ws;
    _Float16* Aexp = (_Float16*)ws;                                        // NN*NN f16   = 2 MB
    _Float16* Bexp = (_Float16*)(ws + (size_t)NN * NN * sizeof(_Float16)); // BATCH*NN f16 = 512 KB
    float* lse = (float*)(ws + (size_t)NN * NN * sizeof(_Float16)
                             + (size_t)BATCH * NN * sizeof(_Float16));     // NN f32
    float* Ma  = lse + NN;                                                 // NN f32
    float* Mb  = Ma + NN;                                                  // BATCH f32

    // 1) column logsumexp of W
    col_lse_kernel<<<NN / 256, 256, 0, stream>>>(W, lse);
    // 2) row max + exp-pack:  logT rows -> Aexp/Ma ; log_alpha rows -> Bexp/Mb
    rowpack_kernel<<<NN, 256, 0, stream>>>(W, lse, 1, Ma, Aexp);
    rowpack_kernel<<<BATCH, 256, 0, stream>>>(log_alpha, nullptr, 0, Mb, Bexp);
    // 3) f16 WMMA GEMM + fused log epilogue
    const int waves = (BATCH / 16) * (NN / 64);   // 256 waves
    logmm_wmma_kernel<<<waves / 4, 128, 0, stream>>>(Aexp, Bexp, Ma, Mb, out);
}
